// Attention_79362405695952
// MI455X (gfx1250) — matrix operand order repaired and checked by reading, not compiled, not run on hardware
//
#include <hip/hip_runtime.h>

// Problem constants (match reference setup_inputs()).
#define B_    512
#define T_    64
#define DIN_  512
#define H_    512
#define C_    96
#define S_    26          // MAXLEN + 1
#define G4H_  2048        // 4*H
#define KCAT_ 1024        // DIN + H (fused [context|h] GEMM K)
#define WIHLD 608         // W_ih row stride (DIN + C)

typedef __attribute__((ext_vector_type(16))) __bf16 v16bf;
typedef __attribute__((ext_vector_type(8)))  __bf16 v8bf;
typedef __attribute__((ext_vector_type(8)))  float  v8f;

// ---------------------------------------------------------------- helpers ---
__device__ __forceinline__ __bf16 f2bf(float f) {
  union { float f; unsigned u; } in; in.f = f;
  unsigned u = in.u + 0x7FFFu + ((in.u >> 16) & 1u);   // round-to-nearest-even
  union { unsigned short s; __bf16 b; } out;
  out.s = (unsigned short)(u >> 16);
  return out.b;
}

__device__ __forceinline__ float sigmoidf_(float x) {
  return 1.0f / (1.0f + __expf(-x));
}

// A fragment: 16(M) x 32(K), bf16, row-major source with leading dim ld.
// ISA 7.12.2 wave32 layout: lane<16 -> row M=lane, halves {K0..7, K16..23};
// lane>=16 -> row M=lane-16, halves {K8..15, K24..31}.
__device__ __forceinline__ v16bf load_frag_mk(const __bf16* __restrict__ base,
                                              int ld, int row0, int k0, int rmax) {
  int lane = threadIdx.x & 31;
  int r = row0 + (lane & 15);
  if (r > rmax) r = rmax;                       // clamp (stores are predicated)
  int kb = (lane & 16) ? 8 : 0;
  const __bf16* q = base + (size_t)r * ld + k0 + kb;
  v8bf lo = *(const v8bf*)(q);                  // K kb..kb+7      (16B load)
  v8bf hi = *(const v8bf*)(q + 16);             // K kb+16..kb+23  (16B load)
  v16bf v;
#pragma unroll
  for (int i = 0; i < 8; ++i) { v[i] = lo[i]; v[i + 8] = hi[i]; }
  return v;
}

// B fragment: 32(K) x 16(N). We compute A @ W^T with W stored [N,K] row-major,
// so the N-row of W is K-contiguous. lane<16 -> col N=lane, K {0..7, 16..23};
// lane>=16 -> col N=lane-16, K {8..15, 24..31} (matches A fragment).
__device__ __forceinline__ v16bf load_frag_nk(const __bf16* __restrict__ base,
                                              int ld, int col0, int k0, int cmax) {
  int lane = threadIdx.x & 31;
  int c = col0 + (lane & 15);
  if (c > cmax) c = cmax;
  int kb = (lane & 16) ? 8 : 0;
  const __bf16* q = base + (size_t)c * ld + k0 + kb;
  v8bf lo = *(const v8bf*)(q);
  v8bf hi = *(const v8bf*)(q + 16);
  v16bf v;
#pragma unroll
  for (int i = 0; i < 8; ++i) { v[i] = lo[i]; v[i + 8] = hi[i]; }
  return v;
}

__device__ __forceinline__ v8f wmma_bf16(v16bf a, v16bf b, v8f c) {
  return __builtin_amdgcn_wmma_f32_16x16x32_bf16(
      /*neg_a=*/false, a, /*neg_b=*/false, b,
      /*c_mod=*/(short)0, c, /*reuse_a=*/false, /*reuse_b=*/false);
}

// C/D 16x16 f32 tile store, predicated on M/N bounds, optional bias.
__device__ __forceinline__ void store_tile(const v8f& acc,
                                           float* __restrict__ Cm, int ldc,
                                           int m0, int n0,
                                           const float* __restrict__ bias,
                                           int M, int N) {
  int lane = threadIdx.x & 31;
  int col  = n0 + (lane & 15);
  if (col >= N) return;
  int rowb = m0 + ((lane & 16) ? 8 : 0);
  float bv = bias ? bias[col] : 0.0f;
#pragma unroll
  for (int v = 0; v < 8; ++v) {
    int r = rowb + v;
    if (r < M) Cm[(size_t)r * ldc + col] = acc[v] + bv;
  }
}

// ------------------------------------------------------------- GEMM kernel ---
// C[M,N] = A[M,K] @ W[N,K]^T (+bias). bf16 operands, f32 accumulate.
// Block = 256 threads = 8 waves arranged 2(M) x 4(N); wave tile = 32x32
// (2x2 WMMA fragments -> 2x operand register reuse). Operands are L2-resident
// (192MB L2 holds the full working set), so no LDS staging: direct global
// fetch + global_prefetch_b8 of the next K slice keeps the serial chain short.
__global__ __launch_bounds__(256)
void gemm_xwt_bf16(const __bf16* __restrict__ A, int lda,
                   const __bf16* __restrict__ W, int ldw,
                   float* __restrict__ Cm, int ldc,
                   const float* __restrict__ bias,
                   int M, int N, int K) {
  int waveId = threadIdx.x >> 5;
  int lane   = threadIdx.x & 31;
  int wm = waveId & 1;
  int wn = waveId >> 1;
  int m0 = blockIdx.y * 64 + wm * 32;
  int n0 = blockIdx.x * 128 + wn * 32;

  v8f acc00 = {}, acc01 = {}, acc10 = {}, acc11 = {};

  for (int k0 = 0; k0 < K; k0 += 32) {
    if (k0 + 32 < K) {   // prefetch next K slice (global_prefetch_b8)
      __builtin_prefetch(A + (size_t)(min(m0 + (lane & 15), M - 1)) * lda + k0 + 32, 0, 1);
      __builtin_prefetch(W + (size_t)(min(n0 + (lane & 15), N - 1)) * ldw + k0 + 32, 0, 1);
    }
    v16bf a0 = load_frag_mk(A, lda, m0,      k0, M - 1);
    v16bf a1 = load_frag_mk(A, lda, m0 + 16, k0, M - 1);
    v16bf b0 = load_frag_nk(W, ldw, n0,      k0, N - 1);
    v16bf b1 = load_frag_nk(W, ldw, n0 + 16, k0, N - 1);
    acc00 = wmma_bf16(a0, b0, acc00);
    acc01 = wmma_bf16(a0, b1, acc01);
    acc10 = wmma_bf16(a1, b0, acc10);
    acc11 = wmma_bf16(a1, b1, acc11);
  }
  store_tile(acc00, Cm, ldc, m0,      n0,      bias, M, N);
  store_tile(acc01, Cm, ldc, m0,      n0 + 16, bias, M, N);
  store_tile(acc10, Cm, ldc, m0 + 16, n0,      bias, M, N);
  store_tile(acc11, Cm, ldc, m0 + 16, n0 + 16, bias, M, N);
}

// ------------------------------------------------------- prep / cast kernels --
__global__ __launch_bounds__(256)
void cast_f32_to_bf16(const float* __restrict__ src, __bf16* __restrict__ dst, long n) {
  long i = (long)blockIdx.x * blockDim.x + threadIdx.x;
  if (i < n) dst[i] = f2bf(src[i]);
}

// Wcat[n, 0:512] = W_ih[n, 0:512]; Wcat[n, 512:1024] = W_hh[n, :]; bcat = b_ih+b_hh
__global__ __launch_bounds__(256)
void build_wcat(const float* __restrict__ W_ih, const float* __restrict__ W_hh,
                const float* __restrict__ b_ih, const float* __restrict__ b_hh,
                __bf16* __restrict__ Wcat, float* __restrict__ bcat) {
  int idx = blockIdx.x * blockDim.x + threadIdx.x;
  if (idx >= G4H_ * KCAT_) return;
  int n = idx >> 10;
  int k = idx & (KCAT_ - 1);
  float v = (k < DIN_) ? W_ih[(size_t)n * WIHLD + k] : W_hh[(size_t)n * H_ + (k - DIN_)];
  Wcat[idx] = f2bf(v);
  if (k == 0) bcat[n] = b_ih[n] + b_hh[n];
}

// h0 = 0 (bf16 half of xcat), c0 = 0
__global__ __launch_bounds__(256)
void init_state(float* __restrict__ c_state, __bf16* __restrict__ xcat) {
  int idx = blockIdx.x * blockDim.x + threadIdx.x;
  if (idx >= B_ * H_) return;
  c_state[idx] = 0.0f;
  int b = idx >> 9, h = idx & (H_ - 1);
  xcat[(size_t)b * KCAT_ + DIN_ + h] = f2bf(0.0f);
}

// --------------------------------------------------- fused attention kernel --
// One block per batch row: e_t = sum_h tanh(H_proj[b,t,h] + hp[b,h])*score[h],
// softmax over T in LDS, context[b,d] = sum_t alpha_t * batch_H[b,t,d].
// Writes context (bf16) into xcat[b, 0:512].
__global__ __launch_bounds__(256)
void attn_step(const float* __restrict__ H_proj,   // [B*T, H]
               const float* __restrict__ batch_H,  // [B, T, DIN]
               const float* __restrict__ hp,       // [B, H]
               const float* __restrict__ score_W,  // [H]
               __bf16* __restrict__ xcat) {        // [B, KCAT]
  __shared__ float hp_s[H_];
  __shared__ float w_s[H_];
  __shared__ float e_s[T_];

  const int b    = blockIdx.x;
  const int tid  = threadIdx.x;
  const int wave = tid >> 5;
  const int lane = tid & 31;

  for (int i = tid; i < H_; i += 256) {
    hp_s[i] = hp[(size_t)b * H_ + i];
    w_s[i]  = score_W[i];
  }
  __syncthreads();

  // scores: 8 waves cover T=64, each wave reduces over H=512 with shfl_xor
  for (int t = wave; t < T_; t += 8) {
    const float* hprow = H_proj + ((size_t)b * T_ + t) * H_;
    float p = 0.0f;
    for (int h = lane; h < H_; h += 32)
      p += tanhf(hprow[h] + hp_s[h]) * w_s[h];
#pragma unroll
    for (int off = 16; off > 0; off >>= 1)
      p += __shfl_xor(p, off, 32);
    if (lane == 0) e_s[t] = p;
  }
  __syncthreads();

  // softmax over T (redundant per-thread reduction over 64 values is cheap)
  float mx = -3.4e38f;
  for (int t = 0; t < T_; ++t) mx = fmaxf(mx, e_s[t]);
  float sum = 0.0f;
  for (int t = 0; t < T_; ++t) sum += __expf(e_s[t] - mx);
  float inv = 1.0f / sum;
  __syncthreads();
  if (tid < T_) e_s[tid] = __expf(e_s[tid] - mx) * inv;   // alpha in place
  __syncthreads();

  // context over batch_H (coalesced along d)
  for (int d = tid; d < DIN_; d += 256) {
    const float* bh = batch_H + (size_t)b * T_ * DIN_ + d;
    float acc = 0.0f;
#pragma unroll 4
    for (int t = 0; t < T_; ++t) acc += e_s[t] * bh[(size_t)t * DIN_];
    xcat[(size_t)b * KCAT_ + d] = f2bf(acc);
  }
}

// ----------------------------------------------------- LSTM elementwise step --
// gates were computed without bias / one-hot; fold bcat and the single one-hot
// column W_ih[:, 512+token] here (the one-hot GEMM reduces to a column gather).
__global__ __launch_bounds__(256)
void lstm_step(const float* __restrict__ gates,   // [B, 4H] raw GEMM out
               const float* __restrict__ bcat,    // [4H]
               const float* __restrict__ W_ih,    // [4H, 608] f32 original
               const int*   __restrict__ text,    // [B, S]
               int s,
               float* __restrict__ c_state,       // [B, H]
               __bf16* __restrict__ xcat,         // [B, KCAT] (h half)
               __bf16* __restrict__ hs) {         // [B, S, H]
  int idx = blockIdx.x * blockDim.x + threadIdx.x;
  if (idx >= B_ * H_) return;
  int b = idx >> 9, h = idx & (H_ - 1);
  int tok = text[(size_t)b * S_ + s];
  const float* g = gates + (size_t)b * G4H_;
  int oc = DIN_ + tok;
  float ig = g[h]            + bcat[h]            + W_ih[(size_t)(h)            * WIHLD + oc];
  float fg = g[H_ + h]       + bcat[H_ + h]       + W_ih[(size_t)(H_ + h)       * WIHLD + oc];
  float gg = g[2 * H_ + h]   + bcat[2 * H_ + h]   + W_ih[(size_t)(2 * H_ + h)   * WIHLD + oc];
  float og = g[3 * H_ + h]   + bcat[3 * H_ + h]   + W_ih[(size_t)(3 * H_ + h)   * WIHLD + oc];
  float cn = sigmoidf_(fg) * c_state[idx] + sigmoidf_(ig) * tanhf(gg);
  float hn = sigmoidf_(og) * tanhf(cn);
  c_state[idx] = cn;
  __bf16 hb = f2bf(hn);
  xcat[(size_t)b * KCAT_ + DIN_ + h] = hb;
  hs[((size_t)b * S_ + s) * H_ + h]  = hb;
}

// ------------------------------------------------------------------ launch ---
extern "C" void kernel_launch(void* const* d_in, const int* in_sizes, int n_in,
                              void* d_out, int out_size, void* d_ws, size_t ws_size,
                              hipStream_t stream) {
  const float* batch_H = (const float*)d_in[0];
  const int*   text    = (const int*)  d_in[1];
  // d_in[2] overlap, d_in[3] scene: unused by reference math
  const float* i2h_W   = (const float*)d_in[4];
  const float* h2h_W   = (const float*)d_in[5];
  const float* h2h_b   = (const float*)d_in[6];
  const float* score_W = (const float*)d_in[7];
  const float* W_ih    = (const float*)d_in[8];
  const float* b_ih    = (const float*)d_in[9];
  const float* W_hh    = (const float*)d_in[10];
  const float* b_hh    = (const float*)d_in[11];
  const float* gen_W   = (const float*)d_in[12];
  const float* gen_b   = (const float*)d_in[13];
  float* out = (float*)d_out;
  (void)in_sizes; (void)n_in; (void)out_size; (void)ws_size;

  char* base = (char*)d_ws;
  size_t off = 0;
  auto carve = [&](size_t bytes) -> void* {
    void* r = base + off;
    off = (off + bytes + 255) & ~(size_t)255;
    return r;
  };
  float*  H_proj  = (float*) carve((size_t)B_ * T_ * H_   * 4);  // 64 MB
  __bf16* bH_bf   = (__bf16*)carve((size_t)B_ * T_ * DIN_ * 2);  // 32 MB
  __bf16* i2h_bf  = (__bf16*)carve((size_t)H_ * DIN_ * 2);
  __bf16* h2h_bf  = (__bf16*)carve((size_t)H_ * H_   * 2);
  __bf16* wcat    = (__bf16*)carve((size_t)G4H_ * KCAT_ * 2);    // 4 MB
  float*  bcat    = (float*) carve((size_t)G4H_ * 4);
  __bf16* genW_bf = (__bf16*)carve((size_t)C_ * H_ * 2);
  __bf16* xcat    = (__bf16*)carve((size_t)B_ * KCAT_ * 2);      // [context|h]
  float*  hp      = (float*) carve((size_t)B_ * H_ * 4);
  float*  gates   = (float*) carve((size_t)B_ * G4H_ * 4);
  float*  c_state = (float*) carve((size_t)B_ * H_ * 4);
  __bf16* hs      = (__bf16*)carve((size_t)B_ * S_ * H_ * 2);    // 13.3 MB

  auto cast = [&](const float* s, __bf16* d, long n) {
    cast_f32_to_bf16<<<dim3((unsigned)((n + 255) / 256)), 256, 0, stream>>>(s, d, n);
  };
  auto gemm = [&](const __bf16* A, int lda, const __bf16* W, int ldw,
                  float* Cm, int ldc, const float* bias, int M, int N, int K) {
    dim3 g((N + 127) / 128, (M + 63) / 64);
    gemm_xwt_bf16<<<g, 256, 0, stream>>>(A, lda, W, ldw, Cm, ldc, bias, M, N, K);
  };

  // One-time prep
  cast(batch_H, bH_bf, (long)B_ * T_ * DIN_);
  cast(i2h_W,   i2h_bf, (long)H_ * DIN_);
  cast(h2h_W,   h2h_bf, (long)H_ * H_);
  cast(gen_W,   genW_bf, (long)C_ * H_);
  build_wcat<<<dim3((G4H_ * KCAT_ + 255) / 256), 256, 0, stream>>>(
      W_ih, W_hh, b_ih, b_hh, wcat, bcat);
  init_state<<<dim3((B_ * H_ + 255) / 256), 256, 0, stream>>>(c_state, xcat);

  // Hoisted projection: H_proj[b*T+t, h] = batch_H[b,t,:] . i2h_W[h,:]
  gemm(bH_bf, DIN_, i2h_bf, DIN_, H_proj, H_, nullptr, B_ * T_, H_, DIN_);

  // 26-step decode scan
  for (int s = 0; s < S_; ++s) {
    // hp = h @ h2h_W^T + h2h_b  (h is the bf16 back half of xcat, lda=KCAT)
    gemm(xcat + DIN_, KCAT_, h2h_bf, H_, hp, H_, h2h_b, B_, H_, H_);
    // fused attention -> context into xcat[:, 0:512]
    attn_step<<<dim3(B_), 256, 0, stream>>>(H_proj, batch_H, hp, score_W, xcat);
    // gates = [context|h] @ [W_ih_ctx|W_hh]^T   (bias/one-hot folded into lstm)
    gemm(xcat, KCAT_, wcat, KCAT_, gates, G4H_, nullptr, B_, G4H_, KCAT_);
    // LSTM elementwise, write h into xcat back half + hs[b,s,:]
    lstm_step<<<dim3((B_ * H_ + 255) / 256), 256, 0, stream>>>(
        gates, bcat, W_ih, text, s, c_state, xcat, hs);
  }

  // logits: out[b,s,:] = hs[b,s,:] @ gen_W^T + gen_b   (M = B*S rows match out)
  gemm(hs, H_, genW_bf, H_, out, C_, gen_b, B_ * S_, C_, H_);
}
